// Transformer_41970420416532
// MI455X (gfx1250) — compile-verified, matrix operand order checked by reading, not run on hardware
//
#include <hip/hip_runtime.h>
#include <hip/hip_bf16.h>
#include <math.h>

// ---------------- problem constants ----------------
constexpr int cB = 2, cS = 1024, cD = 512, cH = 8, cF = 2048, cL = 6, cV = 32000;
constexpr int cDH = cD / cH;                 // 64
constexpr float cEPS = 1e-6f;
constexpr float cEMB_SCALE = 22.62741699796952f;   // sqrt(512)
constexpr float cATT_SCALE = 0.125f;               // 1/sqrt(64)

// ---------------- vector types ----------------
typedef __attribute__((ext_vector_type(16))) _Float16 v16h;
typedef __attribute__((ext_vector_type(8)))  _Float16 v8h;
typedef __attribute__((ext_vector_type(4)))  _Float16 v4h;
typedef __attribute__((ext_vector_type(8)))  float    v8f;

union F16x16 { v16h v; v8h h[2]; };

// =====================================================================
// Generic batched GEMM:  C[z] = A[z] (M,K) * B[z] (K,N or N,K) + bias
// f32 in/out, f16 WMMA compute with f32 accumulate.
// Block: 256 threads = 8 waves; block tile 128x64, K step 32.
// Wave w: mQuad = w&3 (32-row strip), nHalf = w>>2 (32-col strip)
//   -> 2 A-frags x 2 B-frags -> 4 WMMAs per K-step, 32 per block.
// Staging: float4 global loads -> f16 convert -> b64 LDS stores.
// Optional epilogue: + bias[n], + resid (same layout as C), ReLU.
// Batch index z = blockIdx.z ; b = z / Hd, h = z % Hd.
// =====================================================================
constexpr int LSTR = 40;   // LDS row stride in halves (80B rows keep b128/b64 aligned)

__global__ __launch_bounds__(256) void gemm_f16wmma(
    const float* __restrict__ A,  int lda, long sAb, long sAh,
    const float* __restrict__ Bm, int ldb, long sBb, long sBh, int transB,
    const float* __restrict__ bias, const float* __restrict__ resid,
    float* __restrict__ C, int ldc, long sCb, long sCh,
    int M, int N, int K, int relu, int Hd)
{
    __shared__ __align__(16) _Float16 As[128 * LSTR];
    __shared__ __align__(16) _Float16 Bs[64 * LSTR];

    const int z  = blockIdx.z;
    const int bb = z / Hd, hh = z % Hd;
    const float* Ab = A  + (long)bb * sAb + (long)hh * sAh;
    const float* Bb = Bm + (long)bb * sBb + (long)hh * sBh;
    float*       Cb = C  + (long)bb * sCb + (long)hh * sCh;
    const float* Rb = resid ? resid + (long)bb * sCb + (long)hh * sCh : nullptr;

    const int mBase = blockIdx.y * 128;
    const int nBase = blockIdx.x * 64;
    const int tid   = threadIdx.x;
    const int lane  = tid & 31;
    const int wave  = tid >> 5;
    const int mQuad = wave & 3;             // 0..3 : 32-row strip
    const int nHalf = wave >> 2;            // 0..1 : 32-col strip
    const int laneHalf = lane >> 4;         // 0/1
    const int laneRow  = lane & 15;

    v8f acc00 = {}, acc01 = {}, acc10 = {}, acc11 = {};

    for (int kb = 0; kb < K; kb += 32) {
        // ---- issue all global loads for this K-step up front ----
        float4 ar[4];
        #pragma unroll
        for (int p = 0; p < 4; ++p) {
            int idx = tid + p * 256;            // 1024 float4s = 128x32 tile
            int row = idx >> 3, cg = idx & 7;
            ar[p] = *(const float4*)&Ab[(long)(mBase + row) * lda + kb + cg * 4];
        }
        float4 br[2];
        if (transB) {      // B is [N,K] (Q*K^T path): 64x32 tile, rows along K
            #pragma unroll
            for (int p = 0; p < 2; ++p) {
                int idx = tid + p * 256;        // 512 float4s = 64x32 tile
                int n = idx >> 3, cg = idx & 7;
                br[p] = *(const float4*)&Bb[(long)(nBase + n) * ldb + kb + cg * 4];
            }
        } else {           // B is [K,N]: 32x64 tile, coalesced along N
            #pragma unroll
            for (int p = 0; p < 2; ++p) {
                int idx = tid + p * 256;        // 512 float4s = 32x64 tile
                int kk = idx >> 4, ng = idx & 15;
                br[p] = *(const float4*)&Bb[(long)(kb + kk) * ldb + nBase + ng * 4];
            }
        }

        __syncthreads();   // previous iteration's fragment reads are complete

        // ---- convert + store to LDS ----
        #pragma unroll
        for (int p = 0; p < 4; ++p) {
            int idx = tid + p * 256;
            int row = idx >> 3, cg = idx & 7;
            v4h hv = { (_Float16)ar[p].x, (_Float16)ar[p].y,
                       (_Float16)ar[p].z, (_Float16)ar[p].w };
            *(v4h*)&As[row * LSTR + cg * 4] = hv;
        }
        if (transB) {
            #pragma unroll
            for (int p = 0; p < 2; ++p) {
                int idx = tid + p * 256;
                int n = idx >> 3, cg = idx & 7;
                v4h hv = { (_Float16)br[p].x, (_Float16)br[p].y,
                           (_Float16)br[p].z, (_Float16)br[p].w };
                *(v4h*)&Bs[n * LSTR + cg * 4] = hv;
            }
        } else {           // transpose scatter: Bs[n][k]
            #pragma unroll
            for (int p = 0; p < 2; ++p) {
                int idx = tid + p * 256;
                int kk = idx >> 4, ng = idx & 15;
                Bs[(ng * 4 + 0) * LSTR + kk] = (_Float16)br[p].x;
                Bs[(ng * 4 + 1) * LSTR + kk] = (_Float16)br[p].y;
                Bs[(ng * 4 + 2) * LSTR + kk] = (_Float16)br[p].z;
                Bs[(ng * 4 + 3) * LSTR + kk] = (_Float16)br[p].w;
            }
        }
        __syncthreads();

        // ---- fragments per ISA 16-bit layout ----
        // lanes 0-15: K {0..7, 16..23}; lanes 16-31: K {8..15, 24..31}
        F16x16 a0, a1, b0, b1;
        const v8h* pa0 = (const v8h*)&As[((mQuad * 2 + 0) * 16 + laneRow) * LSTR + laneHalf * 8];
        a0.h[0] = pa0[0]; a0.h[1] = pa0[2];
        const v8h* pa1 = (const v8h*)&As[((mQuad * 2 + 1) * 16 + laneRow) * LSTR + laneHalf * 8];
        a1.h[0] = pa1[0]; a1.h[1] = pa1[2];
        const v8h* pb0 = (const v8h*)&Bs[((nHalf * 2 + 0) * 16 + laneRow) * LSTR + laneHalf * 8];
        b0.h[0] = pb0[0]; b0.h[1] = pb0[2];
        const v8h* pb1 = (const v8h*)&Bs[((nHalf * 2 + 1) * 16 + laneRow) * LSTR + laneHalf * 8];
        b1.h[0] = pb1[0]; b1.h[1] = pb1[2];

        acc00 = __builtin_amdgcn_wmma_f32_16x16x32_f16(false, a0.v, false, b0.v,
                                                       (short)0, acc00, false, false);
        acc01 = __builtin_amdgcn_wmma_f32_16x16x32_f16(false, a0.v, false, b1.v,
                                                       (short)0, acc01, false, false);
        acc10 = __builtin_amdgcn_wmma_f32_16x16x32_f16(false, a1.v, false, b0.v,
                                                       (short)0, acc10, false, false);
        acc11 = __builtin_amdgcn_wmma_f32_16x16x32_f16(false, a1.v, false, b1.v,
                                                       (short)0, acc11, false, false);
    }

    // ---- epilogue: C layout lane0-15 N=lane M=r ; lane16-31 N=lane-16 M=r+8 ----
    #pragma unroll
    for (int mj = 0; mj < 2; ++mj) {
        #pragma unroll
        for (int nj = 0; nj < 2; ++nj) {
            v8f acc = mj ? (nj ? acc11 : acc10) : (nj ? acc01 : acc00);
            int n = nBase + (nHalf * 2 + nj) * 16 + laneRow;
            float bv = bias ? bias[n] : 0.0f;
            #pragma unroll
            for (int r = 0; r < 8; ++r) {
                int m = mBase + (mQuad * 2 + mj) * 16 + r + 8 * laneHalf;
                long off = (long)m * ldc + n;
                float val = acc[r] + bv;
                if (Rb) val += Rb[off];
                if (relu) val = val > 0.0f ? val : 0.0f;
                Cb[off] = val;
            }
        }
    }
}

// =====================================================================
// Embedding gather * sqrt(D) + sinusoidal positional encoding
// =====================================================================
__global__ __launch_bounds__(256) void embed_pe_kernel(
    const int* __restrict__ tok, const float* __restrict__ emb,
    float* __restrict__ out)
{
    long i = (long)blockIdx.x * 256 + threadIdx.x;
    if (i >= (long)cB * cS * cD) return;
    int d = (int)(i % cD);
    long bs = i / cD;
    int s = (int)(bs % cS);
    int t = tok[bs];
    float div = expf(-(float)(2 * (d >> 1)) * (logf(10000.0f) / (float)cD));
    float ang = (float)s * div;
    float pe  = (d & 1) ? cosf(ang) : sinf(ang);
    out[i] = emb[(long)t * cD + d] * cEMB_SCALE + pe;
}

// =====================================================================
// LayerNorm, ddof=1, eps added to std:  (x - m) / (std + eps)
// One 256-thread block per row of D=512.
// =====================================================================
__global__ __launch_bounds__(256) void layernorm_kernel(
    const float* __restrict__ x, float* __restrict__ out)
{
    __shared__ float red[256];
    long row = blockIdx.x;
    const float* xr = x + row * cD;
    float* orow = out + row * cD;
    int t = threadIdx.x;
    float v0 = xr[t], v1 = xr[t + 256];
    red[t] = v0 + v1;
    __syncthreads();
    for (int o = 128; o > 0; o >>= 1) { if (t < o) red[t] += red[t + o]; __syncthreads(); }
    float mean = red[0] / (float)cD;
    __syncthreads();
    float d0 = v0 - mean, d1 = v1 - mean;
    red[t] = d0 * d0 + d1 * d1;
    __syncthreads();
    for (int o = 128; o > 0; o >>= 1) { if (t < o) red[t] += red[t + o]; __syncthreads(); }
    float var = red[0] / (float)(cD - 1);
    float inv = 1.0f / (sqrtf(var) + cEPS);
    orow[t]       = d0 * inv;
    orow[t + 256] = d1 * inv;
}

// =====================================================================
// Masked scaled softmax over rows of scores [B,H,S,S].
// mask2d=0: mask is [B,1,1,S]; mask2d=1: mask is [B,1,S,S].
// One 256-thread block per row (S=1024 -> 4 elems/thread).
// =====================================================================
__global__ __launch_bounds__(256) void softmax_mask_kernel(
    float* __restrict__ sc, const int* __restrict__ mask, int mask2d, float scale)
{
    __shared__ float red[256];
    long r = blockIdx.x;                     // 0 .. B*H*S-1
    int b  = (int)(r / ((long)cH * cS));
    int sq = (int)(r % cS);
    float* row = sc + r * (long)cS;
    const int* m = mask2d ? mask + (long)b * cS * cS + (long)sq * cS
                          : mask + (long)b * cS;
    int t = threadIdx.x;
    float vals[4];
    float mx = -3.0e38f;
    #pragma unroll
    for (int j = 0; j < 4; ++j) {
        int c = t + j * 256;
        float v = row[c] * scale;
        if (m[c] == 0) v = -1.0e9f;
        vals[j] = v;
        mx = fmaxf(mx, v);
    }
    red[t] = mx; __syncthreads();
    for (int o = 128; o > 0; o >>= 1) { if (t < o) red[t] = fmaxf(red[t], red[t + o]); __syncthreads(); }
    mx = red[0]; __syncthreads();
    float sum = 0.0f;
    #pragma unroll
    for (int j = 0; j < 4; ++j) { float e = expf(vals[j] - mx); vals[j] = e; sum += e; }
    red[t] = sum; __syncthreads();
    for (int o = 128; o > 0; o >>= 1) { if (t < o) red[t] += red[t + o]; __syncthreads(); }
    float inv = 1.0f / red[0];
    #pragma unroll
    for (int j = 0; j < 4; ++j) row[t + j * 256] = vals[j] * inv;
}

// =====================================================================
// In-place log-softmax over rows of V=32000 (125 elems/thread).
// =====================================================================
__global__ __launch_bounds__(256) void logsoftmax_kernel(float* __restrict__ x)
{
    __shared__ float red[256];
    long r = blockIdx.x;
    float* row = x + r * (long)cV;
    int t = threadIdx.x;
    float mx = -3.0e38f;
    for (int c = t; c < cV; c += 256) mx = fmaxf(mx, row[c]);
    red[t] = mx; __syncthreads();
    for (int o = 128; o > 0; o >>= 1) { if (t < o) red[t] = fmaxf(red[t], red[t + o]); __syncthreads(); }
    mx = red[0]; __syncthreads();
    float sum = 0.0f;
    for (int c = t; c < cV; c += 256) sum += expf(row[c] - mx);
    red[t] = sum; __syncthreads();
    for (int o = 128; o > 0; o >>= 1) { if (t < o) red[t] += red[t + o]; __syncthreads(); }
    float lse = mx + logf(red[0]);
    for (int c = t; c < cV; c += 256) row[c] = row[c] - lse;
}

// =====================================================================
// Host orchestration
// =====================================================================
extern "C" void kernel_launch(void* const* d_in, const int* in_sizes, int n_in,
                              void* d_out, int out_size, void* d_ws, size_t ws_size,
                              hipStream_t stream)
{
    const int*   src      = (const int*)  d_in[0];
    const int*   tgt      = (const int*)  d_in[1];
    const int*   src_mask = (const int*)  d_in[2];
    const int*   tgt_mask = (const int*)  d_in[3];
    const float* src_emb  = (const float*)d_in[4];
    const float* tgt_emb  = (const float*)d_in[5];
    const float* enc_wq = (const float*)d_in[6],  *enc_bq = (const float*)d_in[7];
    const float* enc_wk = (const float*)d_in[8],  *enc_bk = (const float*)d_in[9];
    const float* enc_wv = (const float*)d_in[10], *enc_bv = (const float*)d_in[11];
    const float* enc_wo = (const float*)d_in[12], *enc_bo = (const float*)d_in[13];
    const float* enc_w1 = (const float*)d_in[14], *enc_b1 = (const float*)d_in[15];
    const float* enc_w2 = (const float*)d_in[16], *enc_b2 = (const float*)d_in[17];
    const float* dsa_wq = (const float*)d_in[18], *dsa_bq = (const float*)d_in[19];
    const float* dsa_wk = (const float*)d_in[20], *dsa_bk = (const float*)d_in[21];
    const float* dsa_wv = (const float*)d_in[22], *dsa_bv = (const float*)d_in[23];
    const float* dsa_wo = (const float*)d_in[24], *dsa_bo = (const float*)d_in[25];
    const float* dca_wq = (const float*)d_in[26], *dca_bq = (const float*)d_in[27];
    const float* dca_wk = (const float*)d_in[28], *dca_bk = (const float*)d_in[29];
    const float* dca_wv = (const float*)d_in[30], *dca_bv = (const float*)d_in[31];
    const float* dca_wo = (const float*)d_in[32], *dca_bo = (const float*)d_in[33];
    const float* dec_w1 = (const float*)d_in[34], *dec_b1 = (const float*)d_in[35];
    const float* dec_w2 = (const float*)d_in[36], *dec_b2 = (const float*)d_in[37];
    const float* proj_w = (const float*)d_in[38], *proj_b = (const float*)d_in[39];
    float* out = (float*)d_out;

    const int M  = cB * cS;                   // 2048 token rows
    const size_t XD = (size_t)M * cD;         // 1,048,576 floats
    float* ws  = (float*)d_ws;
    float* x   = ws;                          // activations (x for enc, y for dec)
    float* h   = ws + 1 * XD;                 // layernorm output
    float* q   = ws + 2 * XD;
    float* k   = ws + 3 * XD;
    float* v   = ws + 4 * XD;
    float* ao  = ws + 5 * XD;                 // attention head output (concat)
    float* enc = ws + 6 * XD;                 // encoder final output
    float* ffb = ws + 7 * XD;                 // FFN intermediate (4*XD)
    float* sc  = ws + 11 * XD;                // scores [B,H,S,S] (16*XD)

    auto gemm = [&](const float* A, int lda, long sAb, long sAh,
                    const float* Bm, int ldb, long sBb, long sBh, int tB,
                    const float* bias, const float* resid,
                    float* C, int ldc, long sCb, long sCh,
                    int Mm, int Nn, int Kk, int relu, int nb, int Hd) {
        dim3 grid(Nn / 64, Mm / 128, nb);
        gemm_f16wmma<<<grid, 256, 0, stream>>>(A, lda, sAb, sAh, Bm, ldb, sBb, sBh, tB,
                                               bias, resid, C, ldc, sCb, sCh,
                                               Mm, Nn, Kk, relu, Hd);
    };

    auto ln = [&](const float* in, float* o) {
        layernorm_kernel<<<M, 256, 0, stream>>>(in, o);
    };

    // full MHA: q from xq, k/v from xkv, output += residual written into xio
    auto mha = [&](const float* xq, const float* xkv,
                   const float* wq, const float* bq, const float* wk, const float* bk,
                   const float* wv, const float* bv, const float* wo, const float* bo,
                   const int* mask, int m2d, float* xio) {
        gemm(xq,  cD, 0, 0, wq, cD, 0, 0, 0, bq, nullptr, q, cD, 0, 0, M, cD, cD, 0, 1, 1);
        gemm(xkv, cD, 0, 0, wk, cD, 0, 0, 0, bk, nullptr, k, cD, 0, 0, M, cD, cD, 0, 1, 1);
        gemm(xkv, cD, 0, 0, wv, cD, 0, 0, 0, bv, nullptr, v, cD, 0, 0, M, cD, cD, 0, 1, 1);
        // scores[b,h] = Q K^T   (per-head batched, transB)
        gemm(q, cD, (long)cS * cD, cDH,
             k, cD, (long)cS * cD, cDH, 1,
             nullptr, nullptr,
             sc, cS, (long)cH * cS * cS, (long)cS * cS,
             cS, cS, cDH, 0, cB * cH, cH);
        softmax_mask_kernel<<<cB * cH * cS, 256, 0, stream>>>(sc, mask, m2d, cATT_SCALE);
        // attn @ V -> ao
        gemm(sc, cS, (long)cH * cS * cS, (long)cS * cS,
             v, cD, (long)cS * cD, cDH, 0,
             nullptr, nullptr,
             ao, cD, (long)cS * cD, cDH,
             cS, cDH, cS, 0, cB * cH, cH);
        // output projection + residual (in place on xio)
        gemm(ao, cD, 0, 0, wo, cD, 0, 0, 0, bo, xio, xio, cD, 0, 0, M, cD, cD, 0, 1, 1);
    };

    auto ffn = [&](const float* hin, const float* w1, const float* b1,
                   const float* w2, const float* b2, float* xio) {
        gemm(hin, cD, 0, 0, w1, cF, 0, 0, 0, b1, nullptr, ffb, cF, 0, 0, M, cF, cD, 1, 1, 1);
        gemm(ffb, cF, 0, 0, w2, cD, 0, 0, 0, b2, xio, xio, cD, 0, 0, M, cD, cF, 0, 1, 1);
    };

    const long wDD = (long)cD * cD, wDF = (long)cD * cF;

    // ---------------- encoder ----------------
    embed_pe_kernel<<<(int)(XD / 256), 256, 0, stream>>>(src, src_emb, x);
    for (int i = 0; i < cL; ++i) {
        ln(x, h);
        mha(h, h, enc_wq + i * wDD, enc_bq + i * cD, enc_wk + i * wDD, enc_bk + i * cD,
            enc_wv + i * wDD, enc_bv + i * cD, enc_wo + i * wDD, enc_bo + i * cD,
            src_mask, 0, x);
        ln(x, h);
        ffn(h, enc_w1 + i * wDF, enc_b1 + i * cF, enc_w2 + i * wDF, enc_b2 + i * cD, x);
    }
    ln(x, enc);

    // ---------------- decoder ----------------
    embed_pe_kernel<<<(int)(XD / 256), 256, 0, stream>>>(tgt, tgt_emb, x);
    for (int i = 0; i < cL; ++i) {
        ln(x, h);   // self-attn uses src_mask (as written in the reference)
        mha(h, h, dsa_wq + i * wDD, dsa_bq + i * cD, dsa_wk + i * wDD, dsa_bk + i * cD,
            dsa_wv + i * wDD, dsa_bv + i * cD, dsa_wo + i * wDD, dsa_bo + i * cD,
            src_mask, 0, x);
        ln(x, h);   // cross-attn uses tgt_mask (causal S x S)
        mha(h, enc, dca_wq + i * wDD, dca_bq + i * cD, dca_wk + i * wDD, dca_bk + i * cD,
            dca_wv + i * wDD, dca_bv + i * cD, dca_wo + i * wDD, dca_bo + i * cD,
            tgt_mask, 1, x);
        ln(x, h);
        ffn(h, dec_w1 + i * wDF, dec_b1 + i * cF, dec_w2 + i * wDF, dec_b2 + i * cD, x);
    }
    ln(x, h);

    // ---------------- final projection + log_softmax ----------------
    gemm(h, cD, 0, 0, proj_w, cV, 0, 0, 0, proj_b, nullptr, out, cV, 0, 0, M, cV, cD, 0, 1, 1);
    logsoftmax_kernel<<<M, 256, 0, stream>>>(out);
}
